// CUDATernaryExpertModule_4638564680470
// MI455X (gfx1250) — matrix operand order repaired and checked by reading, not compiled, not run on hardware
//
#include <hip/hip_runtime.h>

typedef __attribute__((ext_vector_type(16))) _Float16 v16h;
typedef __attribute__((ext_vector_type(8)))  _Float16 v8h;
typedef __attribute__((ext_vector_type(4)))  _Float16 v4h;
typedef __attribute__((ext_vector_type(8)))  float    v8f;
typedef __attribute__((ext_vector_type(4)))  float    v4f;
typedef __attribute__((ext_vector_type(4)))  int      v4i;

#define HID   2048
#define INTER 5632
#define TOK   4096

constexpr int KC  = 32;        // K chunk per WMMA stage
constexpr int LDT = KC + 8;    // padded LDS row stride (halves); 80B, 16B-aligned
constexpr int BM  = 64;        // block tile rows (tokens)
constexpr int BN  = 128;       // block tile cols (output channels)

#define AS_GLOBAL __attribute__((address_space(1)))
#define AS_LDS    __attribute__((address_space(3)))

// ---- CDNA5 async global->LDS copy (GLOBAL_LOAD_ASYNC_TO_LDS_B128, ASYNCcnt) ----
__device__ __forceinline__ void async_cp_b128(const _Float16* g, _Float16* l) {
#if __has_builtin(__builtin_amdgcn_global_load_async_to_lds_b128)
  __builtin_amdgcn_global_load_async_to_lds_b128(
      (AS_GLOBAL v4i*)g, (AS_LDS v4i*)l, 0, 0);
#else
  asm volatile("global_load_async_to_lds_b128 %0, %1, off"
               :: "v"((unsigned)(size_t)(AS_LDS _Float16*)l), "v"(g)
               : "memory");
#endif
}

template <int N>
__device__ __forceinline__ void wait_asynccnt() {
#if __has_builtin(__builtin_amdgcn_s_wait_asynccnt)
  __builtin_amdgcn_s_wait_asynccnt(N);
#else
  asm volatile("s_wait_asynccnt %0" :: "n"(N) : "memory");
#endif
}

// ---------------- fp32 -> f16 conversion ----------------
__global__ __launch_bounds__(256)
void cvt_f32_f16(const float* __restrict__ in, _Float16* __restrict__ out) {
  size_t i = ((size_t)blockIdx.x * blockDim.x + threadIdx.x) * 4;
  v4f f = *(const v4f*)(in + i);
  v4h h;
  h.x = (_Float16)f.x; h.y = (_Float16)f.y;
  h.z = (_Float16)f.z; h.w = (_Float16)f.w;
  *(v4h*)(out + i) = h;
}

// ---------------- fragment loaders (ISA 7.12.2 layouts) ----------------
__device__ __forceinline__ v16h lds_frag_a(const _Float16* row, int hi) {
  union { v16h v; v8h h[2]; } t;
  t.h[0] = *(const v8h*)(row + hi * 8);
  t.h[1] = *(const v8h*)(row + 16 + hi * 8);
  return t.v;
}
__device__ __forceinline__ v16h lds_frag_b(const _Float16* row, int hi) {
  union { v16h v; v8h h[2]; } t;
  t.h[0] = *(const v8h*)(row + hi * 8);
  t.h[1] = *(const v8h*)(row + 16 + hi * 8);
  return t.v;
}

__device__ __forceinline__ float silu(float g) {
  return g * (1.0f / (1.0f + __expf(-g)));
}

// ---------------- fused gate/up GEMM + SiLU (double-buffered, async staging) ----------------
__global__ __launch_bounds__(256)
void gateup_gemm(const _Float16* __restrict__ xh, const _Float16* __restrict__ gh,
                 const _Float16* __restrict__ uh, const float* __restrict__ gs,
                 const float* __restrict__ us, _Float16* __restrict__ hdn) {
  __shared__ _Float16 sA[2][BM * LDT];
  __shared__ _Float16 sG[2][BN * LDT];
  __shared__ _Float16 sU[2][BN * LDT];

  const int bm   = blockIdx.x;
  const int bn   = blockIdx.y;
  const int tid  = threadIdx.x;
  const int lane = tid & 31;
  const int wave = tid >> 5;
  const int r    = lane & 15;
  const int hi   = lane >> 4;
  const int wm   = (wave & 1) * 32;
  const int wn   = (wave >> 1) * 32;

  const int arow = tid >> 2, acol = (tid & 3) * 8;   // A: 64x32, 1 x b128/thread
  const int brow = tid >> 1, bcol = (tid & 1) * 16;  // B: 128x32, 2 x b128/thread

  const _Float16* xg = xh + (size_t)(bm * BM + arow) * HID + acol;
  const _Float16* gg = gh + (size_t)(bn * BN + brow) * HID + bcol;
  const _Float16* ug = uh + (size_t)(bn * BN + brow) * HID + bcol;

  auto issue = [&](int kt, int buf) {
    async_cp_b128(xg + kt,     &sA[buf][arow * LDT + acol]);
    async_cp_b128(gg + kt,     &sG[buf][brow * LDT + bcol]);
    async_cp_b128(gg + kt + 8, &sG[buf][brow * LDT + bcol + 8]);
    async_cp_b128(ug + kt,     &sU[buf][brow * LDT + bcol]);
    async_cp_b128(ug + kt + 8, &sU[buf][brow * LDT + bcol + 8]);
  };

  v8f accG[2][2] = {};
  v8f accU[2][2] = {};

  constexpr int NS = HID / KC;   // 64 stages
  issue(0, 0);
  for (int s = 0; s < NS; ++s) {
    const int cur = s & 1;
    if (s + 1 < NS) {
      issue((s + 1) * KC, cur ^ 1);
      wait_asynccnt<5>();        // stage s landed; stage s+1 in flight
    } else {
      wait_asynccnt<0>();
    }
    __syncthreads();

    v16h a[2], bG[2], bU[2];
#pragma unroll
    for (int tm = 0; tm < 2; ++tm)
      a[tm] = lds_frag_a(&sA[cur][(wm + tm * 16 + r) * LDT], hi);
#pragma unroll
    for (int tn = 0; tn < 2; ++tn) {
      bG[tn] = lds_frag_b(&sG[cur][(wn + tn * 16 + r) * LDT], hi);
      bU[tn] = lds_frag_b(&sU[cur][(wn + tn * 16 + r) * LDT], hi);
    }
#pragma unroll
    for (int tm = 0; tm < 2; ++tm)
#pragma unroll
      for (int tn = 0; tn < 2; ++tn) {
        accG[tm][tn] = __builtin_amdgcn_wmma_f32_16x16x32_f16(
            false, a[tm], false, bG[tn], (short)0, accG[tm][tn], false, false);
        accU[tm][tn] = __builtin_amdgcn_wmma_f32_16x16x32_f16(
            false, a[tm], false, bU[tn], (short)0, accU[tm][tn], false, false);
      }
    __syncthreads();             // all waves done reading buf[cur] before re-fill
  }

  // epilogue: scales + SiLU, write hdn (f16)
#pragma unroll
  for (int tn = 0; tn < 2; ++tn) {
    const int i  = bn * BN + wn + tn * 16 + r;
    const float sg = gs[i];
    const float su = us[i];
#pragma unroll
    for (int tm = 0; tm < 2; ++tm) {
      const int t0 = bm * BM + wm + tm * 16 + hi * 8;
#pragma unroll
      for (int v = 0; v < 8; ++v) {
        float g = accG[tm][tn][v] * sg;
        float u = accU[tm][tn][v] * su;
        hdn[(size_t)(t0 + v) * INTER + i] = (_Float16)(silu(g) * u);
      }
    }
  }
}

// ---------------- down GEMM (double-buffered, async staging) ----------------
__global__ __launch_bounds__(256)
void down_gemm(const _Float16* __restrict__ hdn, const _Float16* __restrict__ dh,
               const float* __restrict__ ds, float* __restrict__ out) {
  __shared__ _Float16 sA[2][BM * LDT];
  __shared__ _Float16 sB[2][BN * LDT];

  const int bm   = blockIdx.x;
  const int bn   = blockIdx.y;
  const int tid  = threadIdx.x;
  const int lane = tid & 31;
  const int wave = tid >> 5;
  const int r    = lane & 15;
  const int hi   = lane >> 4;
  const int wm   = (wave & 1) * 32;
  const int wn   = (wave >> 1) * 32;

  const int arow = tid >> 2, acol = (tid & 3) * 8;
  const int brow = tid >> 1, bcol = (tid & 1) * 16;

  const _Float16* ag = hdn + (size_t)(bm * BM + arow) * INTER + acol;
  const _Float16* bg = dh  + (size_t)(bn * BN + brow) * INTER + bcol;

  auto issue = [&](int kt, int buf) {
    async_cp_b128(ag + kt,     &sA[buf][arow * LDT + acol]);
    async_cp_b128(bg + kt,     &sB[buf][brow * LDT + bcol]);
    async_cp_b128(bg + kt + 8, &sB[buf][brow * LDT + bcol + 8]);
  };

  v8f acc[2][2] = {};

  constexpr int NS = INTER / KC;  // 176 stages
  issue(0, 0);
  for (int s = 0; s < NS; ++s) {
    const int cur = s & 1;
    if (s + 1 < NS) {
      issue((s + 1) * KC, cur ^ 1);
      wait_asynccnt<3>();
    } else {
      wait_asynccnt<0>();
    }
    __syncthreads();

    v16h a[2], b[2];
#pragma unroll
    for (int tm = 0; tm < 2; ++tm)
      a[tm] = lds_frag_a(&sA[cur][(wm + tm * 16 + r) * LDT], hi);
#pragma unroll
    for (int tn = 0; tn < 2; ++tn)
      b[tn] = lds_frag_b(&sB[cur][(wn + tn * 16 + r) * LDT], hi);
#pragma unroll
    for (int tm = 0; tm < 2; ++tm)
#pragma unroll
      for (int tn = 0; tn < 2; ++tn)
        acc[tm][tn] = __builtin_amdgcn_wmma_f32_16x16x32_f16(
            false, a[tm], false, b[tn], (short)0, acc[tm][tn], false, false);
    __syncthreads();
  }

#pragma unroll
  for (int tn = 0; tn < 2; ++tn) {
    const int h  = bn * BN + wn + tn * 16 + r;
    const float sd = ds[h];
#pragma unroll
    for (int tm = 0; tm < 2; ++tm) {
      const int t0 = bm * BM + wm + tm * 16 + hi * 8;
#pragma unroll
      for (int v = 0; v < 8; ++v)
        out[(size_t)(t0 + v) * HID + h] = acc[tm][tn][v] * sd;
    }
  }
}

// ---------------- host-side launch ----------------
extern "C" void kernel_launch(void* const* d_in, const int* in_sizes, int n_in,
                              void* d_out, int out_size, void* d_ws, size_t ws_size,
                              hipStream_t stream) {
  (void)in_sizes; (void)n_in; (void)out_size; (void)ws_size;

  const float* x   = (const float*)d_in[0];
  const float* gw  = (const float*)d_in[1];
  const float* uw  = (const float*)d_in[2];
  const float* dw  = (const float*)d_in[3];
  const float* gsc = (const float*)d_in[4];
  const float* usc = (const float*)d_in[5];
  const float* dsc = (const float*)d_in[6];
  float* out = (float*)d_out;

  char* ws = (char*)d_ws;
  const size_t n_x  = (size_t)TOK * HID;
  const size_t n_w  = (size_t)INTER * HID;
  _Float16* xh  = (_Float16*)ws;  ws += n_x * 2;
  _Float16* gh  = (_Float16*)ws;  ws += n_w * 2;
  _Float16* uh  = (_Float16*)ws;  ws += n_w * 2;
  _Float16* dwh = (_Float16*)ws;  ws += n_w * 2;
  _Float16* hdn = (_Float16*)ws;

  cvt_f32_f16<<<(int)(n_x / 4 / 256), 256, 0, stream>>>(x, xh);
  cvt_f32_f16<<<(int)(n_w / 4 / 256), 256, 0, stream>>>(gw, gh);
  cvt_f32_f16<<<(int)(n_w / 4 / 256), 256, 0, stream>>>(uw, uh);
  cvt_f32_f16<<<(int)(n_w / 4 / 256), 256, 0, stream>>>(dw, dwh);

  gateup_gemm<<<dim3(TOK / BM, INTER / BN), 256, 0, stream>>>(xh, gh, uh, gsc, usc, hdn);
  down_gemm<<<dim3(TOK / BM, HID / BN), 256, 0, stream>>>(hdn, dwh, dsc, out);
}
